// features_attention_80616536146399
// MI455X (gfx1250) — matrix operand order repaired and checked by reading, not compiled, not run on hardware
//
#include <hip/hip_runtime.h>
#include <hip/hip_bf16.h>
#include <cstdint>
#include <cstddef>

// ---------------------------------------------------------------------------
// Types for CDNA5 WMMA (wave32): bf16 16x16x32, f32 accumulate
// ---------------------------------------------------------------------------
typedef __attribute__((ext_vector_type(16))) __bf16 v16bf;
typedef __attribute__((ext_vector_type(8)))  float  v8f;
typedef __attribute__((ext_vector_type(4)))  int    v4i;

#define LDS_STRIDE 40   // 32 K-elems + 8 pad (bf16 units); 80B row keeps 16B alignment

static __device__ __forceinline__ unsigned short f32_to_bf16_rne(float f) {
    union { float f; unsigned int u; } v; v.f = f;
    unsigned int u = v.u;
    return (unsigned short)((u + 0x7FFFu + ((u >> 16) & 1u)) >> 16);
}

union FragBF { uint4 q[2]; v16bf v; };

// ---------------------------------------------------------------------------
// CDNA5 async global->LDS copy (GLOBAL_LOAD_ASYNC_TO_LDS_B128, ASYNCcnt).
// Builtin prototype (probe-confirmed by round-2 diagnostic):
//   void __builtin_amdgcn_global_load_async_to_lds_b128(
//        v4i addrspace(1)* gsrc, v4i addrspace(3)* ldst, imm int off, imm int cpol)
// Falls back to VGPR bounce if the toolchain lacks the builtin.
// ---------------------------------------------------------------------------
#if defined(__has_builtin)
#  if __has_builtin(__builtin_amdgcn_global_load_async_to_lds_b128)
#    define HAVE_ASYNC_LDS 1
#  endif
#endif

#ifdef HAVE_ASYNC_LDS
typedef __attribute__((address_space(1))) v4i* as1_v4i_p;
typedef __attribute__((address_space(3))) v4i* as3_v4i_p;
#endif

static __device__ __forceinline__ void async_copy_b128(void* lds_dst, const void* gsrc) {
#ifdef HAVE_ASYNC_LDS
    // generic->AS1: identity address; generic->AS3: low 32 bits are the LDS
    // offset (flat-address truncation rule, ISA 10.2). Cast via integers so
    // the conversions always compile.
    as1_v4i_p g = (as1_v4i_p)(uintptr_t)gsrc;
    as3_v4i_p l = (as3_v4i_p)(unsigned int)(uintptr_t)lds_dst;
    __builtin_amdgcn_global_load_async_to_lds_b128(g, l, 0, 0);
#else
    *(uint4*)lds_dst = *(const uint4*)gsrc;
#endif
}

static __device__ __forceinline__ void async_wait0() {
#ifdef HAVE_ASYNC_LDS
#  if __has_builtin(__builtin_amdgcn_s_wait_asynccnt)
    __builtin_amdgcn_s_wait_asynccnt(0);
#  else
    asm volatile("s_wait_asynccnt 0x0" ::: "memory");
#  endif
#endif
}

// ---------------------------------------------------------------------------
// f32 -> bf16 conversion, 4 elems / thread
// ---------------------------------------------------------------------------
__global__ void __launch_bounds__(256)
cvt_f32_bf16(const float* __restrict__ src, unsigned short* __restrict__ dst, int n4) {
    int i = blockIdx.x * blockDim.x + threadIdx.x;
    if (i >= n4) return;
    float4 f = ((const float4*)src)[i];
    ushort4 h;
    h.x = f32_to_bf16_rne(f.x);
    h.y = f32_to_bf16_rne(f.y);
    h.z = f32_to_bf16_rne(f.z);
    h.w = f32_to_bf16_rne(f.w);
    ((ushort4*)dst)[i] = h;
}

// ---------------------------------------------------------------------------
// NT GEMM: C[M,N] = A[M,K] * B[N,K]^T (+bias).  A,B bf16 K-contiguous rows.
// 128x128 tile / workgroup, 8 waves (4x2), each wave 32x64 = 2x4 WMMA tiles.
// Grids divide exactly -> EXEC always all-ones (WMMA requirement).
// Ch != nullptr -> bf16 output, else f32 to Cf.
// ---------------------------------------------------------------------------
__global__ void __launch_bounds__(256)
gemm_nt_wmma(const unsigned short* __restrict__ A, int lda,
             const unsigned short* __restrict__ Bw, int ldb,
             float* __restrict__ Cf, unsigned short* __restrict__ Ch, int ldc,
             const float* __restrict__ bias, int K) {
    __shared__ unsigned short lA[128 * LDS_STRIDE];
    __shared__ unsigned short lB[128 * LDS_STRIDE];

    const int t    = threadIdx.x;
    const int lane = t & 31;
    const int wid  = t >> 5;
    const int wm   = wid >> 1;     // 0..3 : 32-row band
    const int wn   = wid & 1;      // 0..1 : 64-col band
    const int m0   = blockIdx.x * 128;
    const int n0   = blockIdx.y * 128;

    const v8f vzero = {0.f, 0.f, 0.f, 0.f, 0.f, 0.f, 0.f, 0.f};
    v8f acc[2][4];
    #pragma unroll
    for (int i = 0; i < 2; ++i)
        #pragma unroll
        for (int j = 0; j < 4; ++j) acc[i][j] = vzero;

    const int mf = lane & 15;              // A row / B col / C col within 16
    const int kb = (lane >> 4) * 8;        // A-frag K base (elems)
    const int kh = (lane >> 4) * 16;       // unused; B frag also uses kb

    for (int k0 = 0; k0 < K; k0 += 32) {
        // --- stage A tile: 128 x 32 bf16 (async direct-to-LDS, 2 passes) ---
        #pragma unroll
        for (int p = 0; p < 2; ++p) {
            int idx = (t + p * 256) * 8;
            int r = idx >> 5, c = idx & 31;
            async_copy_b128(&lA[r * LDS_STRIDE + c],
                            &A[(size_t)(m0 + r) * lda + k0 + c]);
        }
        // --- stage B tile: 128 x 32 bf16 (async direct-to-LDS, 2 passes) ---
        #pragma unroll
        for (int p = 0; p < 2; ++p) {
            int idx = (t + p * 256) * 8;
            int r = idx >> 5, c = idx & 31;
            async_copy_b128(&lB[r * LDS_STRIDE + c],
                            &Bw[(size_t)(n0 + r) * ldb + k0 + c]);
        }
        if (k0 + 32 < K) {  // CDNA5 global_prefetch_b8 of next K tiles
            __builtin_prefetch(&A[(size_t)(m0 + (t >> 1)) * lda + k0 + 32], 0, 0);
            __builtin_prefetch(&Bw[(size_t)(n0 + (t >> 1)) * ldb + k0 + 32], 0, 0);
        }
        async_wait0();
        __syncthreads();

        FragBF fa[2], fb[4];
        #pragma unroll
        for (int i = 0; i < 2; ++i) {
            const unsigned short* p0 = &lA[(wm * 32 + i * 16 + mf) * LDS_STRIDE + kb];
            fa[i].q[0] = *(const uint4*)(p0);       // K = kb .. kb+7
            fa[i].q[1] = *(const uint4*)(p0 + 16);  // K = kb+16 .. kb+23
        }
        #pragma unroll
        for (int j = 0; j < 4; ++j) {
            const unsigned short* p0 = &lB[(wn * 64 + j * 16 + mf) * LDS_STRIDE + kb];
            fb[j].q[0] = *(const uint4*)(p0);       // K = kb .. kb+7
            fb[j].q[1] = *(const uint4*)(p0 + 16);  // K = kb+16 .. kb+23
        }
        #pragma unroll
        for (int i = 0; i < 2; ++i)
            #pragma unroll
            for (int j = 0; j < 4; ++j)
                acc[i][j] = __builtin_amdgcn_wmma_f32_16x16x32_bf16(
                    false, fa[i].v, false, fb[j].v, (short)0, acc[i][j], false, false);
        __syncthreads();
    }

    // --- store: C/D layout col = lane&15, row base = (lane>=16)*8 ---
    const int r0 = (lane >> 4) * 8;
    #pragma unroll
    for (int i = 0; i < 2; ++i)
        #pragma unroll
        for (int j = 0; j < 4; ++j) {
            int col = n0 + wn * 64 + j * 16 + mf;
            float bv = bias ? bias[col] : 0.0f;
            #pragma unroll
            for (int r = 0; r < 8; ++r) {
                int row = m0 + wm * 32 + i * 16 + r0 + r;
                float val = acc[i][j][r] + bv;
                if (Ch) Ch[(size_t)row * ldc + col] = f32_to_bf16_rne(val);
                else    Cf[(size_t)row * ldc + col] = val;
            }
        }
}

// ---------------------------------------------------------------------------
// NN GEMM for PV: C[M,N] = P[M,K] * V[K,N].  V staged transposed in LDS
// (transpose-on-store requires the VGPR path; P tile uses async-to-LDS).
// ---------------------------------------------------------------------------
__global__ void __launch_bounds__(256)
gemm_pv_wmma(const unsigned short* __restrict__ P, int lda,
             const unsigned short* __restrict__ V, int ldb,
             float* __restrict__ C, int ldc, int K) {
    __shared__ unsigned short lA[128 * LDS_STRIDE];
    __shared__ unsigned short lBt[128 * LDS_STRIDE];   // [n][k] after transpose

    const int t    = threadIdx.x;
    const int lane = t & 31;
    const int wid  = t >> 5;
    const int wm   = wid >> 1;
    const int wn   = wid & 1;
    const int m0   = blockIdx.x * 128;
    const int n0   = blockIdx.y * 128;

    const v8f vzero = {0.f, 0.f, 0.f, 0.f, 0.f, 0.f, 0.f, 0.f};
    v8f acc[2][4];
    #pragma unroll
    for (int i = 0; i < 2; ++i)
        #pragma unroll
        for (int j = 0; j < 4; ++j) acc[i][j] = vzero;

    const int mf = lane & 15;
    const int kb = (lane >> 4) * 8;
    const int kh = (lane >> 4) * 16;

    for (int k0 = 0; k0 < K; k0 += 32) {
        // --- P tile: 128 x 32 bf16, async direct-to-LDS ---
        #pragma unroll
        for (int p = 0; p < 2; ++p) {
            int idx = (t + p * 256) * 8;
            int r = idx >> 5, c = idx & 31;
            async_copy_b128(&lA[r * LDS_STRIDE + c],
                            &P[(size_t)(m0 + r) * lda + k0 + c]);
        }
        // --- V tile: 32 rows (k) x 128 cols (n) -> transposed into lBt[n][k] ---
        #pragma unroll
        for (int h = 0; h < 2; ++h) {
            int id = t + h * 256;           // 0..511 chunks of 8 cols
            int m  = id >> 4;               // 0..31 : k row
            int e0 = (id & 15) * 8;         // 0..120 : n col base
            uint4 d = *(const uint4*)&V[(size_t)(k0 + m) * ldb + n0 + e0];
            const unsigned short* ds = (const unsigned short*)&d;
            #pragma unroll
            for (int q = 0; q < 8; ++q)
                lBt[(e0 + q) * LDS_STRIDE + m] = ds[q];
        }
        if (k0 + 32 < K) {
            __builtin_prefetch(&P[(size_t)(m0 + (t >> 1)) * lda + k0 + 32], 0, 0);
            __builtin_prefetch(&V[(size_t)(k0 + 32 + (t >> 3)) * ldb + n0], 0, 0);
        }
        async_wait0();
        __syncthreads();

        FragBF fa[2], fb[4];
        #pragma unroll
        for (int i = 0; i < 2; ++i) {
            const unsigned short* p0 = &lA[(wm * 32 + i * 16 + mf) * LDS_STRIDE + kb];
            fa[i].q[0] = *(const uint4*)(p0);
            fa[i].q[1] = *(const uint4*)(p0 + 16);
        }
        #pragma unroll
        for (int j = 0; j < 4; ++j) {
            const unsigned short* p0 = &lBt[(wn * 64 + j * 16 + mf) * LDS_STRIDE + kb];
            fb[j].q[0] = *(const uint4*)(p0);
            fb[j].q[1] = *(const uint4*)(p0 + 16);
        }
        #pragma unroll
        for (int i = 0; i < 2; ++i)
            #pragma unroll
            for (int j = 0; j < 4; ++j)
                acc[i][j] = __builtin_amdgcn_wmma_f32_16x16x32_bf16(
                    false, fa[i].v, false, fb[j].v, (short)0, acc[i][j], false, false);
        __syncthreads();
    }

    const int r0 = (lane >> 4) * 8;
    #pragma unroll
    for (int i = 0; i < 2; ++i)
        #pragma unroll
        for (int j = 0; j < 4; ++j) {
            int col = n0 + wn * 64 + j * 16 + mf;
            #pragma unroll
            for (int r = 0; r < 8; ++r) {
                int row = m0 + wm * 32 + i * 16 + r0 + r;
                C[(size_t)row * ldc + col] = acc[i][j][r];
            }
        }
}

// ---------------------------------------------------------------------------
// Row softmax: one 256-thread workgroup per row of 2048 f32 -> bf16 probs
// ---------------------------------------------------------------------------
__global__ void __launch_bounds__(256)
softmax_rows(const float* __restrict__ S, unsigned short* __restrict__ P, int ncols) {
    __shared__ float red[256];
    const int row = blockIdx.x, t = threadIdx.x;
    const float* src = S + (size_t)row * ncols;

    float v[8];
    float m = -3.4e38f;
    #pragma unroll
    for (int i = 0; i < 8; ++i) { v[i] = src[t + i * 256]; m = fmaxf(m, v[i]); }
    red[t] = m; __syncthreads();
    #pragma unroll
    for (int s = 128; s > 0; s >>= 1) {
        if (t < s) red[t] = fmaxf(red[t], red[t + s]);
        __syncthreads();
    }
    const float rowmax = red[0];
    __syncthreads();

    float sum = 0.f;
    #pragma unroll
    for (int i = 0; i < 8; ++i) { v[i] = __expf(v[i] - rowmax); sum += v[i]; }
    red[t] = sum; __syncthreads();
    #pragma unroll
    for (int s = 128; s > 0; s >>= 1) {
        if (t < s) red[t] += red[t + s];
        __syncthreads();
    }
    const float inv = 1.0f / red[0];

    unsigned short* dst = P + (size_t)row * ncols;
    #pragma unroll
    for (int i = 0; i < 8; ++i)
        dst[t + i * 256] = f32_to_bf16_rne(v[i] * inv);
}

// ---------------------------------------------------------------------------
// Host orchestration (graph-capture safe: only kernel launches on `stream`)
// ---------------------------------------------------------------------------
extern "C" void kernel_launch(void* const* d_in, const int* in_sizes, int n_in,
                              void* d_out, int out_size, void* d_ws, size_t ws_size,
                              hipStream_t stream) {
    const float* x_main = (const float*)d_in[0];
    const float* x_feat = (const float*)d_in[1];
    const float* Wq = (const float*)d_in[2];
    const float* bq = (const float*)d_in[3];
    const float* Wk = (const float*)d_in[4];
    const float* bk = (const float*)d_in[5];
    const float* Wv = (const float*)d_in[6];
    const float* bv = (const float*)d_in[7];
    float* out = (float*)d_out;

    const int Bn = 8, N1 = 2048, N2 = 2048, D = 1024;
    const size_t E1 = (size_t)Bn * N1 * D;   // 16,777,216
    const size_t E2 = (size_t)Bn * N2 * D;
    const size_t EW = (size_t)D * D;

    char* ws = (char*)d_ws;
    size_t off = 0;
    unsigned short* Xm  = (unsigned short*)(ws + off); off += E1 * 2;
    unsigned short* Xf  = (unsigned short*)(ws + off); off += E2 * 2;
    unsigned short* Wqb = (unsigned short*)(ws + off); off += EW * 2;
    unsigned short* Wkb = (unsigned short*)(ws + off); off += EW * 2;
    unsigned short* Wvb = (unsigned short*)(ws + off); off += EW * 2;
    unsigned short* Qb  = (unsigned short*)(ws + off); off += E1 * 2;
    unsigned short* Kb  = (unsigned short*)(ws + off); off += E2 * 2;
    unsigned short* Vb  = (unsigned short*)(ws + off); off += E2 * 2;
    float*          Sb  = (float*)(ws + off);          off += (size_t)N1 * N2 * 4;
    unsigned short* Pb  = (unsigned short*)(ws + off); off += (size_t)N1 * N2 * 2;

    // --- f32 -> bf16 conversions ---
    cvt_f32_bf16<<<(int)(E1 / 4 / 256), 256, 0, stream>>>(x_main, Xm, (int)(E1 / 4));
    cvt_f32_bf16<<<(int)(E2 / 4 / 256), 256, 0, stream>>>(x_feat, Xf, (int)(E2 / 4));
    cvt_f32_bf16<<<(int)(EW / 4 / 256), 256, 0, stream>>>(Wq, Wqb, (int)(EW / 4));
    cvt_f32_bf16<<<(int)(EW / 4 / 256), 256, 0, stream>>>(Wk, Wkb, (int)(EW / 4));
    cvt_f32_bf16<<<(int)(EW / 4 / 256), 256, 0, stream>>>(Wv, Wvb, (int)(EW / 4));

    // --- projections: Y = X @ W^T + b, bf16 out (M = B*N flattened) ---
    dim3 gproj((Bn * N1) / 128, D / 128);
    gemm_nt_wmma<<<gproj, 256, 0, stream>>>(Xm, D, Wqb, D, nullptr, Qb, D, bq, D);
    gemm_nt_wmma<<<gproj, 256, 0, stream>>>(Xf, D, Wkb, D, nullptr, Kb, D, bk, D);
    gemm_nt_wmma<<<gproj, 256, 0, stream>>>(Xf, D, Wvb, D, nullptr, Vb, D, bv, D);

    // --- per-batch attention; S (16.8 MB) stays resident in the 192 MB L2 ---
    dim3 gsc(N1 / 128, N2 / 128);
    dim3 gpv(N1 / 128, D / 128);
    for (int b = 0; b < Bn; ++b) {
        const unsigned short* Qs = Qb + (size_t)b * N1 * D;
        const unsigned short* Ks = Kb + (size_t)b * N2 * D;
        const unsigned short* Vs = Vb + (size_t)b * N2 * D;
        float* Os = out + (size_t)b * N1 * D;

        gemm_nt_wmma<<<gsc, 256, 0, stream>>>(Qs, D, Ks, D, Sb, nullptr, N2, nullptr, D);
        softmax_rows<<<N1, 256, 0, stream>>>(Sb, Pb, N2);
        gemm_pv_wmma<<<gpv, 256, 0, stream>>>(Pb, N2, Vs, D, Os, D, N2);
    }
}
